// TransformerEncoder_21199958573180
// MI455X (gfx1250) — matrix operand order repaired and checked by reading, not compiled, not run on hardware
//
#include <hip/hip_runtime.h>
#include <hip/hip_bf16.h>

#define HIDDEN 768
#define MLP_HIDDEN 3072
#define SEQ 1024
#define BATCH 16
#define HEADS 12
#define HEAD_DIM 64
#define NTOK (BATCH * SEQ) // 16384

typedef __attribute__((ext_vector_type(16))) __bf16 v16bf;
typedef __attribute__((ext_vector_type(8)))  float  v8f;
using u16 = unsigned short;

// ---------------------------------------------------------------------------
// async global->LDS staging (CDNA5 GLOBAL_LOAD_ASYNC_TO_LDS_B128, ASYNCcnt)
// via inline asm: the clang builtin wants LangAS cuda_device int4* params that
// cannot be spelled portably from HIP source.
// LDS address = low 32 bits of flat pointer.
// ---------------------------------------------------------------------------
__device__ __forceinline__ void async_copy16(const u16* g, u16* l) {
#if defined(__HIP_DEVICE_COMPILE__)
  const unsigned lds = (unsigned)(unsigned long long)l;
  asm volatile("global_load_async_to_lds_b128 %0, %1, off"
               :
               : "v"(lds), "v"((unsigned long long)g)
               : "memory");
#else
  *(uint4*)l = *(const uint4*)g;
#endif
}

__device__ __forceinline__ void async_wait_all() {
#if defined(__HIP_DEVICE_COMPILE__)
#if defined(__has_builtin)
#if __has_builtin(__builtin_amdgcn_s_wait_asynccnt)
  __builtin_amdgcn_s_wait_asynccnt(0);
#else
  asm volatile("s_wait_asynccnt 0x0" ::: "memory");
#endif
#else
  asm volatile("s_wait_asynccnt 0x0" ::: "memory");
#endif
#endif
}

// ---------------------------------------------------------------------------
// helpers
// ---------------------------------------------------------------------------
__device__ __forceinline__ u16 f32_to_bf16(float f) {
  unsigned u = __float_as_uint(f);
  u += 0x7FFFu + ((u >> 16) & 1u); // round-to-nearest-even
  return (u16)(u >> 16);
}

// A-fragment: 16x32 bf16 tile, row-major source. ISA layout:
// lanes 0-15 (row = lane): VGPR j<4 -> K=2j,2j+1 ; j>=4 -> K=16+2(j-4)..
// lanes 16-31: same rows, K offset +8 (and +24 for upper half)
__device__ __forceinline__ v16bf load_afrag(const u16* base, int row0, int k0,
                                            int ld, int lane) {
  const int r = lane & 15, hf = lane >> 4;
  const u16* p = base + (row0 + r) * ld + k0 + hf * 8;
  union { v16bf v; unsigned u[8]; } f;
#pragma unroll
  for (int j = 0; j < 8; ++j) {
    const int k = ((j & 4) << 2) + (j & 3) * 2; // j>=4 -> +16
    f.u[j] = *(const unsigned*)(p + k);
  }
  return f.v;
}

// B-fragment: 32x16 bf16 (KxN), loaded from a row-major (N,K) tile.
// lanes 0-15: N=lane, K=0..7,16..23 (2 per VGPR); lanes 16-31: N=lane-16, K=8..15,24..31
__device__ __forceinline__ v16bf load_bfrag(const u16* base, int row0, int k0,
                                            int ld, int lane) {
  const int r = lane & 15, hf = lane >> 4;
  const u16* p = base + (row0 + r) * ld + k0 + hf * 8;
  union { v16bf v; unsigned u[8]; } f;
#pragma unroll
  for (int j = 0; j < 8; ++j) f.u[j] = *(const unsigned*)(p + ((j & 4) << 2) + (j & 3) * 2);
  return f.v;
}

__device__ __forceinline__ v8f wmma_bf16(v16bf a, v16bf b, v8f c) {
  return __builtin_amdgcn_wmma_f32_16x16x32_bf16(
      /*neg_a=*/false, a, /*neg_b=*/false, b,
      /*c_mod=*/(short)0, c, /*reuse_a=*/false, /*reuse_b=*/false);
}

__device__ __forceinline__ void zero_v8f(v8f& v) {
#pragma unroll
  for (int i = 0; i < 8; ++i) v[i] = 0.0f;
}

// ---------------------------------------------------------------------------
// LayerNorm: one row (768) per block, writes bf16
// ---------------------------------------------------------------------------
__global__ __launch_bounds__(256) void ln_kernel(const float* __restrict__ x,
                                                 const float* __restrict__ g,
                                                 const float* __restrict__ b,
                                                 u16* __restrict__ out) {
  const int row = blockIdx.x;
  const float* xr = x + (size_t)row * HIDDEN;
  __shared__ float red[256];
  const int tid = threadIdx.x;
  const float v0 = xr[tid], v1 = xr[tid + 256], v2 = xr[tid + 512];
  red[tid] = v0 + v1 + v2;
  __syncthreads();
  for (int off = 128; off > 0; off >>= 1) {
    if (tid < off) red[tid] += red[tid + off];
    __syncthreads();
  }
  const float mu = red[0] * (1.0f / 768.0f);
  __syncthreads();
  const float d0 = v0 - mu, d1 = v1 - mu, d2 = v2 - mu;
  red[tid] = d0 * d0 + d1 * d1 + d2 * d2;
  __syncthreads();
  for (int off = 128; off > 0; off >>= 1) {
    if (tid < off) red[tid] += red[tid + off];
    __syncthreads();
  }
  const float rstd = rsqrtf(red[0] * (1.0f / 768.0f) + 1e-5f);
  u16* orow = out + (size_t)row * HIDDEN;
  orow[tid]       = f32_to_bf16(d0 * rstd * g[tid]       + b[tid]);
  orow[tid + 256] = f32_to_bf16(d1 * rstd * g[tid + 256] + b[tid + 256]);
  orow[tid + 512] = f32_to_bf16(d2 * rstd * g[tid + 512] + b[tid + 512]);
}

// ---------------------------------------------------------------------------
// Weight conversion: f32 -> bf16 (straight and transposed)
// ---------------------------------------------------------------------------
__global__ void cvt_bf16_kernel(const float* __restrict__ in,
                                u16* __restrict__ out, int n) {
  const int i = blockIdx.x * blockDim.x + threadIdx.x;
  if (i < n) out[i] = f32_to_bf16(in[i]);
}

// in (rows, cols) row-major -> out (cols, rows) row-major, bf16
__global__ void transpose_bf16_kernel(const float* __restrict__ in,
                                      u16* __restrict__ out, int rows,
                                      int cols) {
  const int i = blockIdx.x * blockDim.x + threadIdx.x;
  if (i < rows * cols) {
    const int r = i / cols, c = i - r * cols;
    out[(size_t)c * rows + r] = f32_to_bf16(in[i]);
  }
}

// ---------------------------------------------------------------------------
// Tiled bf16 WMMA GEMM:  C[M,N] = A[M,K] * B[N,K]^T (+ epilogue)
//   MODE 0: store bf16 row-major
//   MODE 1: store bf16 "V-transposed": out[(b*768+n)*1024 + (m&1023)]
//   MODE 2: f32 out = acc + bias[n] + resid[m*N+n]
//   MODE 3: bf16 out = gelu(acc + bias[n])
// Block tile 128x128, BK=32, 8 waves (wave tile 32x64).
// Double-buffered LDS, async global->LDS staging.
// ---------------------------------------------------------------------------
template <int MODE>
__global__ __launch_bounds__(256) void gemm_bf16_kernel(
    const u16* __restrict__ A, const u16* __restrict__ B,
    void* __restrict__ Cout, const float* __restrict__ bias,
    const float* __restrict__ resid, int M, int N, int K) {
  __shared__ u16 As[2][128 * 40];
  __shared__ u16 Bs[2][128 * 40];
  const int tid = threadIdx.x, lane = tid & 31, wave = tid >> 5;
  const int wm = (wave & 3) * 32;  // wave M offset in block tile
  const int wn = (wave >> 2) * 64; // wave N offset in block tile
  const int m0 = blockIdx.y * 128, n0 = blockIdx.x * 128;
  const int lr = tid >> 2, lc = (tid & 3) * 8;

  v8f acc[2][4];
#pragma unroll
  for (int i = 0; i < 2; ++i)
#pragma unroll
    for (int t = 0; t < 4; ++t) zero_v8f(acc[i][t]);

  auto load_tile = [&](int k0, int buf) {
    async_copy16(A + (size_t)(m0 + lr) * K + k0 + lc, &As[buf][lr * 40 + lc]);
    async_copy16(A + (size_t)(m0 + lr + 64) * K + k0 + lc,
                 &As[buf][(lr + 64) * 40 + lc]);
    async_copy16(B + (size_t)(n0 + lr) * K + k0 + lc, &Bs[buf][lr * 40 + lc]);
    async_copy16(B + (size_t)(n0 + lr + 64) * K + k0 + lc,
                 &Bs[buf][(lr + 64) * 40 + lc]);
  };

  load_tile(0, 0);
  async_wait_all();
  __syncthreads();

  int cur = 0;
  for (int k0 = 0; k0 < K; k0 += 32) {
    const bool more = (k0 + 32) < K;
    if (more) load_tile(k0 + 32, cur ^ 1);

    const v16bf af0 = load_afrag(As[cur], wm, 0, 40, lane);
    const v16bf af1 = load_afrag(As[cur], wm + 16, 0, 40, lane);
#pragma unroll
    for (int t = 0; t < 4; ++t) {
      const v16bf bfr = load_bfrag(Bs[cur], wn + t * 16, 0, 40, lane);
      acc[0][t] = wmma_bf16(af0, bfr, acc[0][t]);
      acc[1][t] = wmma_bf16(af1, bfr, acc[1][t]);
    }

    if (more) async_wait_all();
    __syncthreads();
    cur ^= 1;
  }

  const int r = lane & 15, hf = lane >> 4;
#pragma unroll
  for (int mt = 0; mt < 2; ++mt)
#pragma unroll
    for (int t = 0; t < 4; ++t)
#pragma unroll
      for (int v = 0; v < 8; ++v) {
        const int m = m0 + wm + mt * 16 + hf * 8 + v;
        const int n = n0 + wn + t * 16 + r;
        float val = acc[mt][t][v];
        if constexpr (MODE == 0) {
          ((u16*)Cout)[(size_t)m * N + n] = f32_to_bf16(val);
        } else if constexpr (MODE == 1) {
          const int bb = m >> 10; // batch
          const int pos = m & 1023;
          ((u16*)Cout)[((size_t)(bb * HIDDEN + n)) * SEQ + pos] =
              f32_to_bf16(val);
        } else if constexpr (MODE == 2) {
          val += bias[n] + resid[(size_t)m * N + n];
          ((float*)Cout)[(size_t)m * N + n] = val;
        } else {
          val += bias[n];
          val = 0.5f * val * (1.0f + erff(val * 0.70710678118f));
          ((u16*)Cout)[(size_t)m * N + n] = f32_to_bf16(val);
        }
      }
}

// ---------------------------------------------------------------------------
// Flash-style attention. One block = (batch, head, 128-query tile).
// 8 waves; each wave owns 16 query rows. Streams 64-key tiles (async staged).
// q,k: (16384, 768) bf16 row-major. vt: (16, 768, 1024) bf16 (d-major rows).
// out attn: (16384, 768) bf16.
// ---------------------------------------------------------------------------
__global__ __launch_bounds__(256) void attn_kernel(const u16* __restrict__ q,
                                                   const u16* __restrict__ k,
                                                   const u16* __restrict__ vt,
                                                   u16* __restrict__ attn) {
  __shared__ u16 Qs[128 * 72];
  __shared__ u16 Ks[64 * 72];
  __shared__ u16 Vs[64 * 72]; // (d, pos) layout
  __shared__ u16 Ps[128 * 72];
  const int tid = threadIdx.x, lane = tid & 31, wave = tid >> 5;
  const int b = blockIdx.z, h = blockIdx.y, qt = blockIdx.x;
  const int qrow0 = b * SEQ + qt * 128;
  const int col0 = h * HEAD_DIM;
  const int r = lane & 15, hf = lane >> 4;

  // load Q tile 128x64 (async; completion folded into first K/V wait)
#pragma unroll
  for (int i = 0; i < 4; ++i) {
    const int c = tid + 256 * i; // 0..1023 chunks of 8 bf16
    const int row = c >> 3, cc = (c & 7) * 8;
    async_copy16(q + (size_t)(qrow0 + row) * HIDDEN + col0 + cc,
                 &Qs[row * 72 + cc]);
  }

  float rm[8], rs[8];
  v8f o[4];
#pragma unroll
  for (int v = 0; v < 8; ++v) { rm[v] = -1e30f; rs[v] = 0.0f; }
#pragma unroll
  for (int t = 0; t < 4; ++t) zero_v8f(o[t]);

  for (int kt = 0; kt < SEQ / 64; ++kt) {
    __syncthreads(); // previous iteration's readers of Ks/Vs are done
#pragma unroll
    for (int i = 0; i < 2; ++i) {
      const int c = tid + 256 * i; // 0..511
      const int row = c >> 3, cc = (c & 7) * 8;
      async_copy16(k + (size_t)(b * SEQ + kt * 64 + row) * HIDDEN + col0 + cc,
                   &Ks[row * 72 + cc]);
      async_copy16(vt + (size_t)(b * HIDDEN + col0 + row) * SEQ + kt * 64 + cc,
                   &Vs[row * 72 + cc]);
    }
    async_wait_all();
    __syncthreads();

    // S = Q * K^T  (wave: 16 x 64)
    v8f s[4];
#pragma unroll
    for (int t = 0; t < 4; ++t) zero_v8f(s[t]);
#pragma unroll
    for (int ks = 0; ks < 2; ++ks) {
      const v16bf aq = load_afrag(Qs, wave * 16, ks * 32, 72, lane);
#pragma unroll
      for (int t = 0; t < 4; ++t) {
        const v16bf bk = load_bfrag(Ks, t * 16, ks * 32, 72, lane);
        s[t] = wmma_bf16(aq, bk, s[t]);
      }
    }
#pragma unroll
    for (int t = 0; t < 4; ++t)
#pragma unroll
      for (int v = 0; v < 8; ++v) s[t][v] *= 0.125f; // 1/sqrt(64)

    // online softmax: row stats per (half, reg) pair, reduce across 16 lanes
    float tmax[8];
#pragma unroll
    for (int v = 0; v < 8; ++v) {
      float m01 = fmaxf(s[0][v], s[1][v]);
      float m23 = fmaxf(s[2][v], s[3][v]);
      tmax[v] = fmaxf(m01, m23);
      tmax[v] = fmaxf(tmax[v], __shfl_xor(tmax[v], 8, 32));
      tmax[v] = fmaxf(tmax[v], __shfl_xor(tmax[v], 4, 32));
      tmax[v] = fmaxf(tmax[v], __shfl_xor(tmax[v], 2, 32));
      tmax[v] = fmaxf(tmax[v], __shfl_xor(tmax[v], 1, 32));
    }
    float alpha[8];
#pragma unroll
    for (int v = 0; v < 8; ++v) {
      const float nm = fmaxf(rm[v], tmax[v]);
      alpha[v] = __expf(rm[v] - nm);
      rm[v] = nm;
    }
    float psum[8];
#pragma unroll
    for (int v = 0; v < 8; ++v) psum[v] = 0.0f;
#pragma unroll
    for (int t = 0; t < 4; ++t)
#pragma unroll
      for (int v = 0; v < 8; ++v) {
        const float p = __expf(s[t][v] - rm[v]);
        s[t][v] = p;
        psum[v] += p;
      }
#pragma unroll
    for (int v = 0; v < 8; ++v) {
      psum[v] += __shfl_xor(psum[v], 8, 32);
      psum[v] += __shfl_xor(psum[v], 4, 32);
      psum[v] += __shfl_xor(psum[v], 2, 32);
      psum[v] += __shfl_xor(psum[v], 1, 32);
      rs[v] = rs[v] * alpha[v] + psum[v];
    }
#pragma unroll
    for (int t = 0; t < 4; ++t)
#pragma unroll
      for (int v = 0; v < 8; ++v) o[t][v] *= alpha[v];

    // P -> LDS (wave-private rows), then O += P * V^T via WMMA
#pragma unroll
    for (int t = 0; t < 4; ++t)
#pragma unroll
      for (int v = 0; v < 8; ++v)
        Ps[(wave * 16 + hf * 8 + v) * 72 + t * 16 + r] = f32_to_bf16(s[t][v]);
#pragma unroll
    for (int ks = 0; ks < 2; ++ks) {
      const v16bf ap = load_afrag(Ps, wave * 16, ks * 32, 72, lane);
#pragma unroll
      for (int t = 0; t < 4; ++t) {
        const v16bf bv = load_bfrag(Vs, t * 16, ks * 32, 72, lane);
        o[t] = wmma_bf16(ap, bv, o[t]);
      }
    }
  }

  // normalize and store
#pragma unroll
  for (int t = 0; t < 4; ++t)
#pragma unroll
    for (int v = 0; v < 8; ++v) {
      const int row = qrow0 + wave * 16 + hf * 8 + v;
      const int col = col0 + t * 16 + r;
      attn[(size_t)row * HIDDEN + col] = f32_to_bf16(o[t][v] / rs[v]);
    }
}

// ---------------------------------------------------------------------------
// launch
// ---------------------------------------------------------------------------
static constexpr size_t SZ_ACT  = (size_t)NTOK * HIDDEN * 2;
static constexpr size_t SZ_W    = (size_t)HIDDEN * HIDDEN * 2;
static constexpr size_t SZ_W1   = (size_t)HIDDEN * MLP_HIDDEN * 2;
static constexpr size_t SZ_XMID = (size_t)NTOK * HIDDEN * 4;
static constexpr size_t SZ_H    = (size_t)NTOK * MLP_HIDDEN * 2;

static constexpr size_t OFF_XN   = 0;
static constexpr size_t OFF_Q    = OFF_XN + SZ_ACT;
static constexpr size_t OFF_K    = OFF_Q + SZ_ACT;
static constexpr size_t OFF_VT   = OFF_K + SZ_ACT;
static constexpr size_t OFF_ATTN = OFF_VT + SZ_ACT;
static constexpr size_t OFF_XN2  = OFF_ATTN + SZ_ACT;
static constexpr size_t OFF_H    = OFF_XN2 + SZ_ACT;
static constexpr size_t OFF_XMID = OFF_H + SZ_H;
static constexpr size_t OFF_WQ   = OFF_XMID + SZ_XMID;
static constexpr size_t OFF_WK   = OFF_WQ + SZ_W;
static constexpr size_t OFF_WV   = OFF_WK + SZ_W;
static constexpr size_t OFF_WO   = OFF_WV + SZ_W;
static constexpr size_t OFF_W1T  = OFF_WO + SZ_W;
static constexpr size_t OFF_W2T  = OFF_W1T + SZ_W1;

extern "C" void kernel_launch(void* const* d_in, const int* in_sizes, int n_in,
                              void* d_out, int out_size, void* d_ws,
                              size_t ws_size, hipStream_t stream) {
  const float* x     = (const float*)d_in[0];
  const float* ln1_g = (const float*)d_in[1];
  const float* ln1_b = (const float*)d_in[2];
  const float* wq    = (const float*)d_in[3];
  const float* wk    = (const float*)d_in[4];
  const float* wv    = (const float*)d_in[5];
  const float* wo    = (const float*)d_in[6];
  const float* bo    = (const float*)d_in[7];
  const float* ln2_g = (const float*)d_in[8];
  const float* ln2_b = (const float*)d_in[9];
  const float* w1    = (const float*)d_in[10];
  const float* b1    = (const float*)d_in[11];
  const float* w2    = (const float*)d_in[12];
  const float* b2    = (const float*)d_in[13];

  char* ws = (char*)d_ws;
  u16*   xn    = (u16*)(ws + OFF_XN);
  u16*   qb    = (u16*)(ws + OFF_Q);
  u16*   kb    = (u16*)(ws + OFF_K);
  u16*   vtb   = (u16*)(ws + OFF_VT);
  u16*   attnb = (u16*)(ws + OFF_ATTN);
  u16*   xn2   = (u16*)(ws + OFF_XN2);
  u16*   hb    = (u16*)(ws + OFF_H);
  float* xmid  = (float*)(ws + OFF_XMID);
  u16*   wq_bf = (u16*)(ws + OFF_WQ);
  u16*   wk_bf = (u16*)(ws + OFF_WK);
  u16*   wv_bf = (u16*)(ws + OFF_WV);
  u16*   wo_bf = (u16*)(ws + OFF_WO);
  u16*   w1t   = (u16*)(ws + OFF_W1T);
  u16*   w2t   = (u16*)(ws + OFF_W2T);

  const int nW = HIDDEN * HIDDEN;
  const int nW1 = HIDDEN * MLP_HIDDEN;

  // 1) LN1 + weight conversion
  ln_kernel<<<NTOK, 256, 0, stream>>>(x, ln1_g, ln1_b, xn);
  cvt_bf16_kernel<<<(nW + 255) / 256, 256, 0, stream>>>(wq, wq_bf, nW);
  cvt_bf16_kernel<<<(nW + 255) / 256, 256, 0, stream>>>(wk, wk_bf, nW);
  cvt_bf16_kernel<<<(nW + 255) / 256, 256, 0, stream>>>(wv, wv_bf, nW);
  cvt_bf16_kernel<<<(nW + 255) / 256, 256, 0, stream>>>(wo, wo_bf, nW);
  transpose_bf16_kernel<<<(nW1 + 255) / 256, 256, 0, stream>>>(w1, w1t, HIDDEN,
                                                               MLP_HIDDEN);
  transpose_bf16_kernel<<<(nW1 + 255) / 256, 256, 0, stream>>>(w2, w2t,
                                                               MLP_HIDDEN,
                                                               HIDDEN);

  // 2) Q/K/V projections
  const dim3 g768(HIDDEN / 128, NTOK / 128); // (6,128)
  gemm_bf16_kernel<0><<<g768, 256, 0, stream>>>(xn, wq_bf, qb, nullptr, nullptr,
                                                NTOK, HIDDEN, HIDDEN);
  gemm_bf16_kernel<0><<<g768, 256, 0, stream>>>(xn, wk_bf, kb, nullptr, nullptr,
                                                NTOK, HIDDEN, HIDDEN);
  gemm_bf16_kernel<1><<<g768, 256, 0, stream>>>(xn, wv_bf, vtb, nullptr,
                                                nullptr, NTOK, HIDDEN, HIDDEN);

  // 3) attention
  attn_kernel<<<dim3(SEQ / 128, HEADS, BATCH), 256, 0, stream>>>(qb, kb, vtb,
                                                                 attnb);

  // 4) output projection + residual -> xmid (f32)
  gemm_bf16_kernel<2><<<g768, 256, 0, stream>>>(attnb, wo_bf, xmid, bo, x,
                                                NTOK, HIDDEN, HIDDEN);

  // 5) LN2
  ln_kernel<<<NTOK, 256, 0, stream>>>(xmid, ln2_g, ln2_b, xn2);

  // 6) MLP up + GELU -> h (bf16)
  const dim3 g3072(MLP_HIDDEN / 128, NTOK / 128); // (24,128)
  gemm_bf16_kernel<3><<<g3072, 256, 0, stream>>>(xn2, w1t, hb, b1, nullptr,
                                                 NTOK, MLP_HIDDEN, HIDDEN);

  // 7) MLP down + bias + residual -> d_out (f32)
  gemm_bf16_kernel<2><<<g768, 256, 0, stream>>>(hb, w2t, (float*)d_out, b2,
                                                xmid, NTOK, HIDDEN, MLP_HIDDEN);
}
